// RelationalOperatorNet_81733227643464
// MI455X (gfx1250) — matrix operand order repaired and checked by reading, not compiled, not run on hardware
//
#include <hip/hip_runtime.h>
#include <hip/hip_bf16.h>
#include <stdint.h>

#define B_ROWS   32768
#define N_OPC    8
#define N_PAIRC  28
#define RD       64
#define RD2      32
#define FD       128
#define KTOT     2048           // 28*64 + 8*32
#define PAIR_CH  1792           // 28*64
#define EPS_DIV  1e-8f
#define EPS_BN   1e-5f

__constant__ int cII[N_PAIRC] = {0,0,0,0,0,0,0,1,1,1,1,1,1,2,2,2,2,2,3,3,3,3,4,4,4,5,5,6};
__constant__ int cJJ[N_PAIRC] = {1,2,3,4,5,6,7,2,3,4,5,6,7,3,4,5,6,7,4,5,6,7,5,6,7,6,7,7};

typedef __attribute__((ext_vector_type(16))) __bf16 v16bf;
typedef __attribute__((ext_vector_type(8)))  float  v8f;
typedef __attribute__((ext_vector_type(4)))  int    v4i;

typedef __attribute__((address_space(1))) v4i g_v4i;   // global
typedef __attribute__((address_space(3))) v4i l_v4i;   // LDS

// ---- CDNA5 async global->LDS path (ASYNCcnt), guarded with fallbacks ----
#if defined(__has_builtin)
# if __has_builtin(__builtin_amdgcn_global_load_async_to_lds_b128)
#  define USE_ASYNC 1
# endif
#endif
#ifndef USE_ASYNC
# define USE_ASYNC 0
#endif

#if defined(__has_builtin)
# if __has_builtin(__builtin_amdgcn_s_wait_asynccnt)
#  define WAIT_ASYNC(n) __builtin_amdgcn_s_wait_asynccnt(n)
# elif USE_ASYNC
#  define WAIT_ASYNC(n) asm volatile("s_wait_asynccnt %0" ::"i"(n) : "memory")
# else
#  define WAIT_ASYNC(n)
# endif
#else
# define WAIT_ASYNC(n)
#endif

__device__ __forceinline__ void cp16(const char* g, char* l) {
#if USE_ASYNC
  // generic LDS pointer carries the LDS byte address in its low 32 bits
  __builtin_amdgcn_global_load_async_to_lds_b128(
      (g_v4i*)(uintptr_t)g,
      (l_v4i*)(uint32_t)(uintptr_t)l,
      0, 0);
#else
  *(uint4*)l = *(const uint4*)g;
#endif
}

__device__ __forceinline__ unsigned short f2bf(float f) {
  unsigned int u = __float_as_uint(f);
  u += 0x7fffu + ((u >> 16) & 1u);               // round-to-nearest-even
  return (unsigned short)(u >> 16);
}

// ---------------------------------------------------------------- zero stats
__global__ void kzero(float* __restrict__ sums, float* __restrict__ sumsq) {
  int i = blockIdx.x * 256 + threadIdx.x;
  if (i < KTOT) { sums[i] = 0.f; sumsq[i] = 0.f; }
}

// -------------------------------------------- featurize + relu-act + stats
#define ROWS_A 64
__global__ __launch_bounds__(256) void kfeat(
    const float* __restrict__ x,
    const float* __restrict__ pw_W, const float* __restrict__ pw_b,
    const float* __restrict__ rs_W, const float* __restrict__ rs_b,
    unsigned short* __restrict__ act,
    float* __restrict__ sums, float* __restrict__ sumsq)
{
  __shared__ float feat[ROWS_A][120];   // 116 used, padded
  const int tid  = threadIdx.x;
  const int row0 = blockIdx.x * ROWS_A;

  for (int idx = tid; idx < ROWS_A * 116; idx += 256) {
    int r = idx / 116, e = idx - r * 116;
    const float* xr = x + (size_t)(row0 + r) * 9;
    float val;
    if (e < 84) {                       // pairwise: (p, q)
      int p = e / 3, q = e - p * 3;
      float oi = xr[cII[p]], oj = xr[cJJ[p]];
      val = (q == 0) ? (oi - oj) : (q == 1) ? (oi / (oj + EPS_DIV)) : (oi * oj);
    } else {                            // result features: (n, q)
      int e2 = e - 84; int n = e2 >> 2, q = e2 & 3;
      float res = xr[8], o = xr[n];
      val = (q == 0) ? (res - o) : (q == 1) ? (res / (o + EPS_DIV))
                                 : (q == 2) ? res : o;
    }
    feat[r][e] = val;
  }
  __syncthreads();

  const int p0 = tid >> 6, d0 = tid & 63;
  float w[7][3], bb[7];
#pragma unroll
  for (int j = 0; j < 7; ++j) {
    int p = p0 + 4 * j;
    w[j][0] = pw_W[(p * 3 + 0) * RD + d0];
    w[j][1] = pw_W[(p * 3 + 1) * RD + d0];
    w[j][2] = pw_W[(p * 3 + 2) * RD + d0];
    bb[j]   = pw_b[p * RD + d0];
  }
  const int n8 = tid >> 5, d8 = tid & 31;
  float w8[4];
#pragma unroll
  for (int q = 0; q < 4; ++q) w8[q] = rs_W[(n8 * 4 + q) * RD2 + d8];
  const float b8 = rs_b[n8 * RD2 + d8];

  float s[8], s2[8];
#pragma unroll
  for (int j = 0; j < 8; ++j) { s[j] = 0.f; s2[j] = 0.f; }

  for (int r = 0; r < ROWS_A; ++r) {
    size_t rowbase = (size_t)(row0 + r) * KTOT;
#pragma unroll
    for (int j = 0; j < 7; ++j) {
      int p = p0 + 4 * j;
      float a = fmaf(feat[r][p * 3 + 2], w[j][2],
               fmaf(feat[r][p * 3 + 1], w[j][1],
               fmaf(feat[r][p * 3 + 0], w[j][0], bb[j])));
      a = fmaxf(a, 0.f);
      act[rowbase + tid + 256 * j] = f2bf(a);
      s[j] += a; s2[j] += a * a;
    }
    const float* fr = &feat[r][84 + n8 * 4];
    float a = fmaf(fr[3], w8[3], fmaf(fr[2], w8[2],
              fmaf(fr[1], w8[1], fmaf(fr[0], w8[0], b8))));
    a = fmaxf(a, 0.f);
    act[rowbase + PAIR_CH + tid] = f2bf(a);
    s[7] += a; s2[7] += a * a;
  }
#pragma unroll
  for (int j = 0; j < 8; ++j) {
    int c = (j < 7) ? (tid + 256 * j) : (PAIR_CH + tid);
    atomicAdd(&sums[c],  s[j]);
    atomicAdd(&sumsq[c], s2[j]);
  }
}

// ---------------------------------------------------------- BN scale/shift
__global__ void kbn(const float* __restrict__ sums, const float* __restrict__ sumsq,
                    const float* __restrict__ pw_gamma, const float* __restrict__ pw_beta,
                    const float* __restrict__ rs_gamma, const float* __restrict__ rs_beta,
                    float* __restrict__ scale, float* __restrict__ shift)
{
  int c = blockIdx.x * 256 + threadIdx.x;
  if (c >= KTOT) return;
  const float invB = 1.f / (float)B_ROWS;
  float mean = sums[c] * invB;
  float var  = sumsq[c] * invB - mean * mean;
  float g, be;
  if (c < PAIR_CH) { g = pw_gamma[c]; be = pw_beta[c]; }
  else             { g = rs_gamma[c - PAIR_CH]; be = rs_beta[c - PAIR_CH]; }
  float sc = g * rsqrtf(var + EPS_BN);
  scale[c] = sc;
  shift[c] = be - mean * sc;
}

// ------------------------------------------- fold BN into FC1; bf16 weights
__global__ __launch_bounds__(256) void kfold(
    const float* __restrict__ f1_W, const float* __restrict__ f1_b,
    const float* __restrict__ f2_W,
    const float* __restrict__ scale, const float* __restrict__ shift,
    unsigned short* __restrict__ Wp, unsigned short* __restrict__ f2h,
    float* __restrict__ b1p)
{
  int b = blockIdx.x;
  if (b < 1024) {                         // Wp = scale[k] * f1_W : 2048x128 bf16
    int idx = b * 256 + threadIdx.x;
    int c = idx >> 7;
    Wp[idx] = f2bf(scale[c] * f1_W[idx]);
  } else if (b < 1056) {                  // f2_W -> bf16 : 128x64
    int idx = (b - 1024) * 256 + threadIdx.x;
    f2h[idx] = f2bf(f2_W[idx]);
  } else {                                // b1' = f1_b + shift @ f1_W
    int n = threadIdx.x;
    if (n < FD) {
      float accv = f1_b[n];
      for (int c = 0; c < KTOT; ++c) accv = fmaf(shift[c], f1_W[c * FD + n], accv);
      b1p[n] = accv;
    }
  }
}

// ------------------------------------------------- fused WMMA GEMM pipeline
#define TM 128
#define NK (KTOT / 32)
__global__ __launch_bounds__(256) void kgemm(
    const unsigned short* __restrict__ act,   // B x 2048 bf16
    const unsigned short* __restrict__ Wp,    // 2048 x 128 bf16 (BN-folded)
    const float* __restrict__ b1p,            // 128
    const unsigned short* __restrict__ f2h,   // 128 x 64 bf16
    const float* __restrict__ f2_b,           // 64
    const float* __restrict__ f3_W,           // 64
    const float* __restrict__ f3_b,           // 1
    float* __restrict__ out)                  // B x 1
{
  // LDS layout (byte offsets), bank-conflict padded row strides:
  //  f2s [0,18432)           128 rows * 144B
  //  red [18432,18944)       128 f32
  //  A0  [18944,29184) 128*80B   B0 [29184,37888) 32*272B    } reused after the
  //  A1  [37888,48128) 128*80B   B1 [48128,56832) 32*272B    } K-loop as
  //  C1  [18944,53760)       128 rows * 272B                 } C1 tile
  __shared__ __align__(16) char smem[56832];
  const int F2S = 0, RED = 18432, C1O = 18944;
  const int AOFS[2] = {18944, 37888};
  const int BOFS[2] = {29184, 48128};

  const int tid  = threadIdx.x;
  const int lane = tid & 31;
  const int wave = tid >> 5;        // 0..7 -> 16-row strip each
  const int half = lane >> 4;
  const int l15  = lane & 15;
  const int m0   = blockIdx.x * TM;
  const int wm0  = wave * 16;

  if (tid < 128) *(float*)(smem + RED + tid * 4) = 0.f;

  // stage f2 weights (16KB) via the async queue; drained by first loop wait
  {
    const char* src = (const char*)f2h;            // 1024 16B chunks
#pragma unroll
    for (int i = 0; i < 4; ++i) {
      int q = tid + i * 256;
      int r = q >> 3, sg = q & 7;
      cp16(src + q * 16, smem + F2S + r * 144 + sg * 16);
    }
  }

  // A-tile staging addresses: 512 chunks (128 rows x 4 segs of 16B)
  const char* actbase = (const char*)act + (size_t)m0 * (KTOT * 2);
  const char* wpbase  = (const char*)Wp;

  // issue K-slice `ks` into LDS buffer `buf`
  auto stage = [&](int buf, int ks) {
#pragma unroll
    for (int i = 0; i < 2; ++i) {                  // A tile: 128 x 64B
      int q = tid + i * 256;
      int r = q >> 2, sg = q & 3;
      cp16(actbase + (size_t)r * (KTOT * 2) + ks * 64 + sg * 16,
           smem + AOFS[buf] + r * 80 + sg * 16);
    }
#pragma unroll
    for (int i = 0; i < 2; ++i) {                  // B tile: 32 x 256B
      int q = tid + i * 256;
      int r = q >> 4, sg = q & 15;
      cp16(wpbase + ((size_t)(ks * 32 + r)) * (FD * 2) + sg * 16,
           smem + BOFS[buf] + r * 272 + sg * 16);
    }
  };

  // ---- GEMM1: (128 x 2048) @ (2048 x 128), C initialized with folded bias
  v8f acc[8];
#pragma unroll
  for (int t = 0; t < 8; ++t) {
    float bv = b1p[t * 16 + l15];
    v8f a8;
#pragma unroll
    for (int i = 0; i < 8; ++i) a8[i] = bv;
    acc[t] = a8;
  }

  union Frag { v16bf v; uint4 q[2]; };

  stage(0, 0);                                     // prologue
  for (int ks = 0; ks < NK; ++ks) {
    const int cur = ks & 1;
    if (ks + 1 < NK) {
      stage(cur ^ 1, ks + 1);                      // prefetch next slice
      WAIT_ASYNC(4);                               // drain current slice only
    } else {
      WAIT_ASYNC(0);                               // drain everything
    }
    __syncthreads();                               // cross-wave visibility

    // A fragment: lane<16 -> K {0..7,16..23}; lane>=16 -> K {8..15,24..31}
    Frag a;
    {
      const char* ar = smem + AOFS[cur] + (wm0 + l15) * 80 + half * 16;
      a.q[0] = *(const uint4*)(ar);
      a.q[1] = *(const uint4*)(ar + 32);
    }
#pragma unroll
    for (int t = 0; t < 8; ++t) {                  // B fragment: lane%16 = N-col, K same order as A
      Frag b;
      const char* br = smem + BOFS[cur] + (t * 16 + l15) * 2;
#pragma unroll
      for (int e = 0; e < 16; ++e) {
        int k = (e < 8) ? (half * 8 + e) : (16 + half * 8 + (e - 8));
        ((unsigned short*)&b)[e] = *(const unsigned short*)(br + k * 272);
      }
      acc[t] = __builtin_amdgcn_wmma_f32_16x16x32_bf16(
          false, a.v, false, b.v, (short)0, acc[t], false, false);
    }
    __syncthreads();       // reads of buf `cur` done -> reusable next iter
  }

  // relu -> bf16 C1 tile (row-major 128 x 128, stride 272B) over dead buffers
#pragma unroll
  for (int t = 0; t < 8; ++t) {
    int n = t * 16 + l15;
#pragma unroll
    for (int r = 0; r < 8; ++r) {
      int m = wm0 + r + 8 * half;
      *(unsigned short*)(smem + C1O + m * 272 + n * 2) = f2bf(fmaxf(acc[t][r], 0.f));
    }
  }
  __syncthreads();

  // ---- GEMM2: C1 (128x128) @ f2 (128x64), C initialized with f2_b
  v8f acc2[4];
#pragma unroll
  for (int t = 0; t < 4; ++t) {
    float bv = f2_b[t * 16 + l15];
    v8f a8;
#pragma unroll
    for (int i = 0; i < 8; ++i) a8[i] = bv;
    acc2[t] = a8;
  }
#pragma unroll
  for (int s = 0; s < 4; ++s) {
    Frag a;
    const char* ar = smem + C1O + (wm0 + l15) * 272 + s * 64 + half * 16;
    a.q[0] = *(const uint4*)(ar);
    a.q[1] = *(const uint4*)(ar + 32);
#pragma unroll
    for (int t = 0; t < 4; ++t) {
      Frag b;
      const char* br = smem + F2S + (s * 32) * 144 + (t * 16 + l15) * 2;
#pragma unroll
      for (int e = 0; e < 16; ++e) {
        int k = (e < 8) ? (half * 8 + e) : (16 + half * 8 + (e - 8));
        ((unsigned short*)&b)[e] = *(const unsigned short*)(br + k * 144);
      }
      acc2[t] = __builtin_amdgcn_wmma_f32_16x16x32_bf16(
          false, a.v, false, b.v, (short)0, acc2[t], false, false);
    }
  }

  // ---- GEMM3: relu(C2) @ f3_W (64x1) via LDS f32 atomics
  float p[8];
#pragma unroll
  for (int r = 0; r < 8; ++r) p[r] = 0.f;
#pragma unroll
  for (int t = 0; t < 4; ++t) {
    float w3 = f3_W[t * 16 + l15];
#pragma unroll
    for (int r = 0; r < 8; ++r)
      p[r] = fmaf(fmaxf(acc2[t][r], 0.f), w3, p[r]);
  }
#pragma unroll
  for (int r = 0; r < 8; ++r) {
    int m = wm0 + r + 8 * half;
    atomicAdd((float*)(smem + RED + m * 4), p[r]);
  }
  __syncthreads();
  if (tid < 128) out[m0 + tid] = *(float*)(smem + RED + tid * 4) + f3_b[0];
}

// ---------------------------------------------------------------- launcher
extern "C" void kernel_launch(void* const* d_in, const int* in_sizes, int n_in,
                              void* d_out, int out_size, void* d_ws, size_t ws_size,
                              hipStream_t stream) {
  const float* x        = (const float*)d_in[0];
  const float* pw_W     = (const float*)d_in[1];
  const float* pw_b     = (const float*)d_in[2];
  const float* pw_gamma = (const float*)d_in[3];
  const float* pw_beta  = (const float*)d_in[4];
  const float* rs_W     = (const float*)d_in[5];
  const float* rs_b     = (const float*)d_in[6];
  const float* rs_gamma = (const float*)d_in[7];
  const float* rs_beta  = (const float*)d_in[8];
  const float* f1_W     = (const float*)d_in[9];
  const float* f1_b     = (const float*)d_in[10];
  const float* f2_W     = (const float*)d_in[11];
  const float* f2_b     = (const float*)d_in[12];
  const float* f3_W     = (const float*)d_in[13];
  const float* f3_b     = (const float*)d_in[14];
  float* out            = (float*)d_out;

  char* ws = (char*)d_ws;
  float*          sums  = (float*)(ws + 0);
  float*          sumsq = (float*)(ws + 8192);
  float*          scale = (float*)(ws + 16384);
  float*          shift = (float*)(ws + 24576);
  float*          b1p   = (float*)(ws + 32768);
  unsigned short* f2h   = (unsigned short*)(ws + 33280);
  unsigned short* Wp    = (unsigned short*)(ws + 49664);
  unsigned short* act   = (unsigned short*)(ws + 573952);   // B x 2048 bf16

  kzero<<<8, 256, 0, stream>>>(sums, sumsq);
  kfeat<<<B_ROWS / ROWS_A, 256, 0, stream>>>(x, pw_W, pw_b, rs_W, rs_b,
                                             act, sums, sumsq);
  kbn<<<8, 256, 0, stream>>>(sums, sumsq, pw_gamma, pw_beta, rs_gamma, rs_beta,
                             scale, shift);
  kfold<<<1057, 256, 0, stream>>>(f1_W, f1_b, f2_W, scale, shift, Wp, f2h, b1p);
  kgemm<<<B_ROWS / TM, 256, 0, stream>>>(act, Wp, b1p, f2h, f2_b, f3_W, f3_b, out);
}
